// ALMTermsClass_68685116998321
// MI455X (gfx1250) — compile-verified, matrix operand order checked by reading, not run on hardware
//
#include <hip/hip_runtime.h>

typedef __attribute__((ext_vector_type(16))) _Float16 v16h;
typedef __attribute__((ext_vector_type(8)))  _Float16 h8;
typedef __attribute__((ext_vector_type(2)))  _Float16 h2;
typedef __attribute__((ext_vector_type(8)))  float    v8f;
typedef int v4i __attribute__((vector_size(16)));   // GCC-style, matches builtin sig

#define ALM_DELTA 0.1f

// Async global->LDS staging (ASYNCcnt path), guarded so a toolchain without
// the builtin still compiles via the plain-copy fallback.
#if defined(__has_builtin)
#if __has_builtin(__builtin_amdgcn_global_load_async_to_lds_b128) && \
    __has_builtin(__builtin_amdgcn_s_wait_asynccnt)
#define ALM_ASYNC_LDS 1
#endif
#if __has_builtin(__builtin_elementwise_maxnum)
#define ALM_VMAX(a, b) __builtin_elementwise_maxnum((a), (b))
#endif
#endif
#ifndef ALM_ASYNC_LDS
#define ALM_ASYNC_LDS 0
#endif
#ifndef ALM_VMAX
#define ALM_VMAX(a, b) __builtin_elementwise_max((a), (b))
#endif

// ---------------------------------------------------------------------------
// Kernel 1: out[0..L) = lambdas, out[L] = 0   (harness poisons d_out)
// ---------------------------------------------------------------------------
__global__ void alm_init_out(const float* __restrict__ lambdas,
                             float* __restrict__ out, int L) {
    int i = blockIdx.x * blockDim.x + threadIdx.x;
    if (i < L) out[i] = lambdas[i];
    if (i == 0) out[L] = 0.0f;
}

// ---------------------------------------------------------------------------
// Kernel 2: q-row-sums via WMMA (f16 margin tile x ones -> f32 acc).
//   grid.x : pos tiles (WAVES tiles of 16 rows per block, one tile per wave)
//   grid.y : K-chunks of KCHUNK neg values (staged in LDS per block)
// Partial q per K-chunk atomically composes: out[idx[i]] = lam[i]+mu*sum(parts)
// ---------------------------------------------------------------------------
template <int WAVES, int KCHUNK>
__global__ __launch_bounds__(WAVES * 32)
void alm_qsum_wmma(const float* __restrict__ pos,
                   const float* __restrict__ neg,
                   const int*   __restrict__ idx,
                   const float* __restrict__ mu_p,
                   float* __restrict__ out,
                   int P, int N) {
    extern __shared__ __align__(16) unsigned char smem[];
    float*    negs32 = reinterpret_cast<float*>(smem);                       // KCHUNK f32
    _Float16* negs16 = reinterpret_cast<_Float16*>(smem + KCHUNK * sizeof(float)); // KCHUNK f16

    const int tid   = threadIdx.x;
    const int lane  = tid & 31;
    const int wave  = tid >> 5;
    const int kbase = blockIdx.y * KCHUNK;

    // ---- Stage f32 neg chunk into LDS ------------------------------------
#if ALM_ASYNC_LDS
    for (int i = tid * 4; i < KCHUNK; i += WAVES * 32 * 4) {
        auto g = (__attribute__((address_space(1))) v4i*)(neg + kbase + i);
        auto l = (__attribute__((address_space(3))) v4i*)(negs32 + i);
        __builtin_amdgcn_global_load_async_to_lds_b128(g, l, 0, 0);
    }
    __builtin_amdgcn_s_wait_asynccnt(0);
#else
    {
        const float4* src = reinterpret_cast<const float4*>(neg + kbase);
        float4* dst = reinterpret_cast<float4*>(negs32);
        for (int i = tid; i < (KCHUNK >> 2); i += WAVES * 32) dst[i] = src[i];
    }
#endif
    __syncthreads();

    // ---- One-time packed f32 -> f16 conversion of the chunk ---------------
    for (int i = tid * 2; i < KCHUNK; i += WAVES * 32 * 2) {
        float2 v = *reinterpret_cast<const float2*>(negs32 + i);
        h2 hv;
        hv.x = (_Float16)v.x;
        hv.y = (_Float16)v.y;
        *reinterpret_cast<h2*>(negs16 + i) = hv;   // v_cvt_pk_f16_f32 + ds_store_b32
    }
    __syncthreads();

    // ---- Per-wave 16-row tile --------------------------------------------
    const int tileRow = (blockIdx.x * WAVES + wave) * 16;
    const int hi      = lane >> 4;                 // lane half -> K sub-ranges
    const float bias  = ALM_DELTA - pos[tileRow + (lane & 15)];
    const _Float16 bh = (_Float16)bias;

    h8 bias8;
#pragma unroll
    for (int i = 0; i < 8; ++i) bias8[i] = bh;
    h8 zero8 = {};

    v16h ones;
#pragma unroll
    for (int i = 0; i < 16; ++i) ones[i] = (_Float16)1.0f;

    v8f acc = {};   // 16x16 f32 accumulator; every column = running row-sum

    // 16-bit A layout (16x32, wave32):
    //   lanes 0-15 : halves K 0..7  and 16..23
    //   lanes16-31 : halves K 8..15 and 24..31
#pragma unroll 2
    for (int kb = 0; kb < KCHUNK; kb += 32) {
        const _Float16* p = negs16 + kb + hi * 8;      // 16B-aligned
        h8 n1 = *reinterpret_cast<const h8*>(p);       // ds_load_b128
        h8 n2 = *reinterpret_cast<const h8*>(p + 16);  // ds_load_b128
        h8 m1 = ALM_VMAX(n1 + bias8, zero8);           // v_pk_add_f16 + v_pk_max_f16
        h8 m2 = ALM_VMAX(n2 + bias8, zero8);
        v16h a = __builtin_shufflevector(m1, m2,
                 0, 1, 2, 3, 4, 5, 6, 7, 8, 9, 10, 11, 12, 13, 14, 15);
        // D = A x ones + C  ->  row sums accumulate in every column, f32.
        acc = __builtin_amdgcn_wmma_f32_16x16x32_f16(
            /*neg_a=*/false, a, /*neg_b=*/false, ones,
            /*c_mod=*/(short)0, acc, /*reuse_a=*/false, /*reuse_b=*/false);
    }

    // C/D layout: lane 0 holds D[M=0..7][N=0] in acc[0..7],
    //             lane 16 holds D[M=8..15][N=0].
    const float mu = mu_p[0];
    if (lane == 0 || lane == 16) {
        const int rbase = tileRow + (lane >> 4) * 8;
#pragma unroll
        for (int r = 0; r < 8; ++r) {
            const int row = rbase + r;
            atomicAdd(&out[idx[row]], mu * acc[r]);
        }
    }
    (void)P; (void)N;
}

// ---------------------------------------------------------------------------
// Kernel 3: finalize loss from fully-accumulated out[] and ORIGINAL lambdas
// (pre-update value; inputs are never mutated).
//   q_last = (out[idx[P-1]] - lambdas[idx[P-1]]) / mu
//   loss   = (mu/2 * q_last^2 + lam_last * q_last) / (P*N)
// ---------------------------------------------------------------------------
__global__ void alm_loss(const int* __restrict__ idx,
                         const float* __restrict__ lambdas,
                         const float* __restrict__ mu_p,
                         float* __restrict__ out,
                         int P, int N, int L) {
    if (threadIdx.x == 0 && blockIdx.x == 0) {
        const int   id  = idx[P - 1];
        const float mu  = mu_p[0];
        const float lam = lambdas[id];
        const float q   = (out[id] - lam) / mu;
        out[L] = (mu * 0.5f * q * q + lam * q) / ((float)P * (float)N);
    }
}

// ---------------------------------------------------------------------------
extern "C" void kernel_launch(void* const* d_in, const int* in_sizes, int n_in,
                              void* d_out, int out_size, void* d_ws, size_t ws_size,
                              hipStream_t stream) {
    const float* pos = (const float*)d_in[0];
    const float* neg = (const float*)d_in[1];
    const int*   idx = (const int*)d_in[2];
    const float* lam = (const float*)d_in[3];
    const float* mu  = (const float*)d_in[4];
    float* out = (float*)d_out;

    const int P = in_sizes[0];   // 4096
    const int N = in_sizes[1];   // 8192
    const int L = in_sizes[3];   // 4096

    // 1) out = [lambdas, 0]
    alm_init_out<<<(L + 255) / 256, 256, 0, stream>>>(lam, out, L);

    // 2) WMMA row-sum + scatter-add partials
    constexpr int WAVES  = 4;     // 128 threads / block, 1 tile per wave
    constexpr int KCHUNK = 2048;  // 8 KB f32 + 4 KB f16 LDS per block
    const int tiles   = P / 16;                  // 256
    const int xblocks = tiles / WAVES;           // 64
    const int yblocks = N / KCHUNK;              // 4
    dim3 grid(xblocks, yblocks, 1);
    const size_t shbytes = KCHUNK * sizeof(float) + KCHUNK * sizeof(_Float16);
    alm_qsum_wmma<WAVES, KCHUNK>
        <<<grid, WAVES * 32, shbytes, stream>>>(pos, neg, idx, mu, out, P, N);

    // 3) loss finalize (stream-ordered after all partials land)
    alm_loss<<<1, 32, 0, stream>>>(idx, lam, mu, out, P, N, L);

    (void)n_in; (void)out_size; (void)d_ws; (void)ws_size;
}
